// InformerBlock_44985487458557
// MI455X (gfx1250) — matrix operand order repaired and checked by reading, not compiled, not run on hardware
//
#include <hip/hip_runtime.h>
#include <hip/hip_bf16.h>

// ---------------------------------------------------------------------------
// Types for CDNA5 WMMA / TDM
// ---------------------------------------------------------------------------
typedef __attribute__((ext_vector_type(16))) __bf16 v16bf;
typedef __attribute__((ext_vector_type(8)))  float  v8f;
typedef unsigned int v4u __attribute__((ext_vector_type(4)));
typedef int          v8i __attribute__((ext_vector_type(8)));
typedef int          v4i __attribute__((ext_vector_type(4)));

static __device__ __forceinline__ unsigned short f2bf_bits(float f) {
    unsigned u = __float_as_uint(f);
    unsigned r = u + 0x7FFFu + ((u >> 16) & 1u);   // round-to-nearest-even
    return (unsigned short)(r >> 16);
}

static __device__ __forceinline__ v8f wmma_bf16(v16bf a, v16bf b, v8f c) {
    return __builtin_amdgcn_wmma_f32_16x16x32_bf16(
        /*neg_a=*/false, a, /*neg_b=*/false, b,
        /*c_mod=*/(short)0, c, /*reuse_a=*/false, /*reuse_b=*/false);
}

// Generic->LDS byte offset: for LDS apertures ADDR[31:0] is the LDS offset.
static __device__ __forceinline__ unsigned lds_byte_off(const void* p) {
    return (unsigned)(unsigned long long)p;
}

// ---------------------------------------------------------------------------
// Tensor Data Mover: 2-D tile load (bf16 elements) Global -> LDS.
// Tile = tile_rows rows of tile_k contiguous elements, row pitch stride_elems.
// LDS receives the tile fully packed ([tile_rows][tile_k]). Tensor dims are
// set equal to the tile (tiles never cross an edge here -> no OOB clipping).
// D# layout per CDNA5 ISA 8.3/8.4. Issued by one wave; EXEC is ignored.
// This toolchain exposes the 6-arg builtin:
//   (uint32x4 g0, int32x8 g1, int32x4 g2, int32x4 g3, int32x8 g4, i32 cpol)
// Groups 2/3/4 are zero (2-D descriptor: tile_dim2/3/4=0, no gather/iterate).
// ---------------------------------------------------------------------------
static __device__ __forceinline__ void tdm_load_2d(unsigned lds_addr,
                                                   const void* gptr,
                                                   unsigned tile_k,
                                                   unsigned tile_rows,
                                                   unsigned long long stride_elems) {
    unsigned long long ga = (unsigned long long)(size_t)gptr;
    v4u g0;
    g0[0] = 1u;                                                  // count=1, user D#
    g0[1] = lds_addr;                                            // lds_addr (bytes)
    g0[2] = (unsigned)ga;                                        // global_addr[31:0]
    g0[3] = (unsigned)((ga >> 32) & 0x01FFFFFFu) | 0x80000000u;  // addr[56:32] | type=2
    v8i g1;
    g1[0] = (int)(1u << 16);                                     // data_size=1 (2B); wg_mask=0
    g1[1] = (int)((tile_k & 0xFFFFu) << 16);                     // tensor_dim0[15:0]
    g1[2] = (int)((tile_k >> 16) | ((tile_rows & 0xFFFFu) << 16)); // d0[31:16] | tensor_dim1[15:0]
    g1[3] = (int)((tile_rows >> 16) | ((tile_k & 0xFFFFu) << 16)); // d1[31:16] | tile_dim0
    g1[4] = (int)(tile_rows & 0xFFFFu);                          // tile_dim1 (tile_dim2=0)
    g1[5] = (int)(unsigned)stride_elems;                         // tensor_dim0_stride[31:0]
    g1[6] = (int)(unsigned)((stride_elems >> 32) & 0xFFFFull);   // stride[47:32]
    g1[7] = 0;
    v4i z4 = {0, 0, 0, 0};
    v8i z8 = {0, 0, 0, 0, 0, 0, 0, 0};
    __builtin_amdgcn_tensor_load_to_lds(g0, g1, z4, z4, z8, 0);
}

// A fragment: 16x32 bf16, source row-major [M][K] with leading dim ld (K even).
// ISA layout: lane<16 -> row=lane, K in {0..7, 16..23}; lane>=16 -> K in {8..15, 24..31}.
static __device__ __forceinline__ v16bf load_frag_a(const unsigned short* src, int ld) {
    const int lane = threadIdx.x & 31;
    const int m = lane & 15, hi = lane >> 4;
    const unsigned short* row = src + (size_t)m * ld;
    union { v16bf v; unsigned u[8]; } f;
#pragma unroll
    for (int i = 0; i < 8; ++i) {
        int k = ((i >> 2) << 4) + hi * 8 + ((i & 3) << 1);
        f.u[i] = *(const unsigned*)(row + k);
    }
    return f.v;
}

// B fragment: 32x16 bf16 given as B^T row-major [N][K] with leading dim ld.
// ISA layout: lane n holds column n; lanes 0..15 -> K in {0..7, 16..23}, lanes 16..31 -> K in {8..15, 24..31}.
static __device__ __forceinline__ v16bf load_frag_b(const unsigned short* srcT, int ld) {
    const int lane = threadIdx.x & 31;
    const int n = lane & 15, hi = lane >> 4;
    const unsigned short* row = srcT + (size_t)n * ld;
    union { v16bf v; unsigned u[8]; } f;
#pragma unroll
    for (int i = 0; i < 8; ++i) {
        int k = ((i >> 2) << 4) + hi * 8 + ((i & 3) << 1);
        f.u[i] = *(const unsigned*)(row + k);
    }
    return f.v;
}

// ---------------------------------------------------------------------------
// Elementwise conversion helpers
// ---------------------------------------------------------------------------
__global__ void cvt_bf16_kernel(const float* __restrict__ src,
                                unsigned short* __restrict__ dst, int n) {
    int i = blockIdx.x * blockDim.x + threadIdx.x;
    if (i < n) dst[i] = f2bf_bits(src[i]);
}

// dst[C][R] = bf16(src[R][C])
__global__ void transpose_cvt_kernel(const float* __restrict__ src,
                                     unsigned short* __restrict__ dst, int R, int C) {
    int i = blockIdx.x * blockDim.x + threadIdx.x;
    if (i < R * C) {
        int r = i / C, c = i - r * C;
        dst[(size_t)c * R + r] = f2bf_bits(src[i]);
    }
}

// ---------------------------------------------------------------------------
// Tiled bf16 GEMM:  C[M][N] = A[M][K] * B (BT[N][K]) + bias.
// Block = 128 threads (4 waves), tile 64x64, K-step 64, TDM-staged LDS tiles.
// EPI 0: QKV epilogue (scatter to per-head q(*8)/k/v-transposed, bf16)
// EPI 1: ReLU, store bf16 [M][N]
// EPI 2: plain f32 store [M][N]
// ---------------------------------------------------------------------------
template <int EPI>
__global__ void __launch_bounds__(128)
gemm_bf16_kernel(const unsigned short* __restrict__ A,
                 const unsigned short* __restrict__ BT,
                 const float* __restrict__ bias,
                 int M, int N, int K,
                 float* __restrict__ outF,
                 unsigned short* __restrict__ outBF,
                 unsigned short* __restrict__ qOut,
                 unsigned short* __restrict__ kOut,
                 unsigned short* __restrict__ vtOut) {
    __shared__ unsigned short As[64 * 64];   // [row][k] 8KB
    __shared__ unsigned short Bs[64 * 64];   // [col][k] 8KB
    (void)M;

    const int tid = threadIdx.x;
    const int wid = tid >> 5, lane = tid & 31;
    const int m0 = blockIdx.x * 64, n0 = blockIdx.y * 64;

    v8f acc[4];
    acc[0] = {}; acc[1] = {}; acc[2] = {}; acc[3] = {};

    for (int k0 = 0; k0 < K; k0 += 64) {
        __syncthreads();  // previous tiles fully consumed before DMA overwrite
        if (wid == 0) {
            tdm_load_2d(lds_byte_off(As), A + (size_t)m0 * K + k0, 64, 64, (unsigned long long)K);
            tdm_load_2d(lds_byte_off(Bs), BT + (size_t)n0 * K + k0, 64, 64, (unsigned long long)K);
            __builtin_amdgcn_s_wait_tensorcnt(0);
        }
        __syncthreads();

        v16bf af0 = load_frag_a(As + wid * 16 * 64, 64);
        v16bf af1 = load_frag_a(As + wid * 16 * 64 + 32, 64);
#pragma unroll
        for (int j = 0; j < 4; ++j) {
            acc[j] = wmma_bf16(af0, load_frag_b(Bs + j * 16 * 64, 64), acc[j]);
            acc[j] = wmma_bf16(af1, load_frag_b(Bs + j * 16 * 64 + 32, 64), acc[j]);
        }
    }

    const int hi = lane >> 4, nl = lane & 15;
#pragma unroll
    for (int j = 0; j < 4; ++j) {
        int c = n0 + j * 16 + nl;
        float bv = bias ? bias[c] : 0.0f;
#pragma unroll
        for (int r = 0; r < 8; ++r) {
            int m = m0 + wid * 16 + r + 8 * hi;
            float val = acc[j][r] + bv;
            if (EPI == 0) {
                // columns: [0,512)=Q (pre-scaled by sqrt(hd)=8), [512,1024)=K, rest=V^T
                int b = m >> 11, l = m & 2047;
                if (c < 512) {
                    int h = c >> 6, e = c & 63;
                    qOut[(((size_t)(b * 8 + h) * 2048 + l) << 6) + e] = f2bf_bits(val * 8.0f);
                } else if (c < 1024) {
                    int cc = c - 512, h = cc >> 6, e = cc & 63;
                    kOut[(((size_t)(b * 8 + h) * 2048 + l) << 6) + e] = f2bf_bits(val);
                } else {
                    int cc = c - 1024, h = cc >> 6, e = cc & 63;
                    vtOut[((((size_t)(b * 8 + h)) << 6) + e) * 2048 + l] = f2bf_bits(val);
                }
            } else if (EPI == 1) {
                float rv = val > 0.0f ? val : 0.0f;
                outBF[(size_t)m * N + c] = f2bf_bits(rv);
            } else {
                outF[(size_t)m * N + c] = val;
            }
        }
    }
}

// ---------------------------------------------------------------------------
// Flash attention. Block = 4 waves = 64 queries of one (b,h); K/V tiles of 64
// keys staged once per block via TDM and shared by all 4 waves. Q was
// pre-scaled by sqrt(hd) so scores = Q'·K^T directly.
// Q/K: [B*H][L][64] bf16; VT: [B*H][64][L] bf16; out: f32 [B][L][512].
// ---------------------------------------------------------------------------
__global__ void __launch_bounds__(128)
attn_kernel(const unsigned short* __restrict__ Q,
            const unsigned short* __restrict__ Kmat,
            const unsigned short* __restrict__ VT,
            float* __restrict__ out) {
    __shared__ unsigned short Ks[64 * 64];     // [key][hd]   8KB
    __shared__ unsigned short Vs[64 * 64];     // [hd][key]   8KB
    __shared__ unsigned short Pt[4][16 * 64];  // per-wave P relayout bounce, 8KB

    const int tid = threadIdx.x, wid = tid >> 5, lane = tid & 31;
    const int hi = lane >> 4, nl = lane & 15;
    const int nqb = 2048 / 64;
    const int bh = blockIdx.x / nqb;
    const int qb = (blockIdx.x % nqb) * 64 + wid * 16;
    const int b = bh >> 3, h = bh & 7;

    const unsigned short* qp  = Q    + ((size_t)bh * 2048 + qb) * 64;
    const unsigned short* kp  = Kmat + (size_t)bh * 2048 * 64;
    const unsigned short* vtp = VT   + (size_t)bh * 64 * 2048;

    v16bf q0 = load_frag_a(qp, 64);
    v16bf q1 = load_frag_a(qp + 32, 64);

    v8f o0 = {}, o1 = {}, o2 = {}, o3 = {};
    float rm[8], rl[8];
#pragma unroll
    for (int r = 0; r < 8; ++r) { rm[r] = -3.4e38f; rl[r] = 0.0f; }

    for (int kb = 0; kb < 2048; kb += 64) {
        __syncthreads();
        if (wid == 0) {
            tdm_load_2d(lds_byte_off(Ks), kp + (size_t)kb * 64, 64, 64, 64ull);
            tdm_load_2d(lds_byte_off(Vs), vtp + kb, 64, 64, 2048ull);
            __builtin_amdgcn_s_wait_tensorcnt(0);
        }
        __syncthreads();

        v8f s0 = {}, s1 = {}, s2 = {}, s3 = {};
        s0 = wmma_bf16(q0, load_frag_b(Ks + 0 * 16 * 64, 64), s0);
        s0 = wmma_bf16(q1, load_frag_b(Ks + 0 * 16 * 64 + 32, 64), s0);
        s1 = wmma_bf16(q0, load_frag_b(Ks + 1 * 16 * 64, 64), s1);
        s1 = wmma_bf16(q1, load_frag_b(Ks + 1 * 16 * 64 + 32, 64), s1);
        s2 = wmma_bf16(q0, load_frag_b(Ks + 2 * 16 * 64, 64), s2);
        s2 = wmma_bf16(q1, load_frag_b(Ks + 2 * 16 * 64 + 32, 64), s2);
        s3 = wmma_bf16(q0, load_frag_b(Ks + 3 * 16 * 64, 64), s3);
        s3 = wmma_bf16(q1, load_frag_b(Ks + 3 * 16 * 64 + 32, 64), s3);

#pragma unroll
        for (int r = 0; r < 8; ++r) {
            float a0 = s0[r], a1 = s1[r], a2 = s2[r], a3 = s3[r];
            float mx = fmaxf(fmaxf(a0, a1), fmaxf(a2, a3));
            mx = fmaxf(mx, __shfl_xor(mx, 1, 32));
            mx = fmaxf(mx, __shfl_xor(mx, 2, 32));
            mx = fmaxf(mx, __shfl_xor(mx, 4, 32));
            mx = fmaxf(mx, __shfl_xor(mx, 8, 32));
            float nm = fmaxf(rm[r], mx);
            float alpha = __expf(rm[r] - nm);
            rm[r] = nm;
            float p0 = __expf(a0 - nm), p1 = __expf(a1 - nm);
            float p2 = __expf(a2 - nm), p3 = __expf(a3 - nm);
            float rs = (p0 + p1) + (p2 + p3);
            rs += __shfl_xor(rs, 1, 32);
            rs += __shfl_xor(rs, 2, 32);
            rs += __shfl_xor(rs, 4, 32);
            rs += __shfl_xor(rs, 8, 32);
            rl[r] = rl[r] * alpha + rs;
            o0[r] *= alpha; o1[r] *= alpha; o2[r] *= alpha; o3[r] *= alpha;
            int rr = r + 8 * hi;
            Pt[wid][rr * 64 + nl]      = f2bf_bits(p0);
            Pt[wid][rr * 64 + 16 + nl] = f2bf_bits(p1);
            Pt[wid][rr * 64 + 32 + nl] = f2bf_bits(p2);
            Pt[wid][rr * 64 + 48 + nl] = f2bf_bits(p3);
        }
        // LDS is in-order per wave: loads below see the stores above.
        v16bf pf0 = load_frag_a(&Pt[wid][0], 64);
        v16bf pf1 = load_frag_a(&Pt[wid][0] + 32, 64);
        o0 = wmma_bf16(pf0, load_frag_b(Vs + 0 * 16 * 64, 64), o0);
        o0 = wmma_bf16(pf1, load_frag_b(Vs + 0 * 16 * 64 + 32, 64), o0);
        o1 = wmma_bf16(pf0, load_frag_b(Vs + 1 * 16 * 64, 64), o1);
        o1 = wmma_bf16(pf1, load_frag_b(Vs + 1 * 16 * 64 + 32, 64), o1);
        o2 = wmma_bf16(pf0, load_frag_b(Vs + 2 * 16 * 64, 64), o2);
        o2 = wmma_bf16(pf1, load_frag_b(Vs + 2 * 16 * 64 + 32, 64), o2);
        o3 = wmma_bf16(pf0, load_frag_b(Vs + 3 * 16 * 64, 64), o3);
        o3 = wmma_bf16(pf1, load_frag_b(Vs + 3 * 16 * 64 + 32, 64), o3);
    }

#pragma unroll
    for (int r = 0; r < 8; ++r) {
        float inv = 1.0f / rl[r];
        int l = qb + r + 8 * hi;
        float* orow = out + ((size_t)b * 2048 + l) * 512 + h * 64;
        orow[nl]      = o0[r] * inv;
        orow[16 + nl] = o1[r] * inv;
        orow[32 + nl] = o2[r] * inv;
        orow[48 + nl] = o3[r] * inv;
    }
}

// ---------------------------------------------------------------------------
// Fused residual + LayerNorm over d=512. One 128-thread block per row.
// ---------------------------------------------------------------------------
__global__ void __launch_bounds__(128)
ln_kernel(const float* __restrict__ xa, const float* __restrict__ xb,
          const float* __restrict__ g, const float* __restrict__ be,
          float* __restrict__ yf, unsigned short* __restrict__ ybf) {
    const int row = blockIdx.x;
    const int tid = threadIdx.x, wid = tid >> 5, lane = tid & 31;
    const float* pa = xa + (size_t)row * 512;
    const float* pb = xb + (size_t)row * 512;

    float v[4], sum = 0.0f, sq = 0.0f;
#pragma unroll
    for (int i = 0; i < 4; ++i) {
        int idx = tid + i * 128;
        float t = pa[idx] + pb[idx];
        v[i] = t; sum += t; sq += t * t;
    }
#pragma unroll
    for (int off = 16; off >= 1; off >>= 1) {
        sum += __shfl_xor(sum, off, 32);
        sq  += __shfl_xor(sq,  off, 32);
    }
    __shared__ float s1[4], s2[4];
    if (lane == 0) { s1[wid] = sum; s2[wid] = sq; }
    __syncthreads();
    sum = s1[0] + s1[1] + s1[2] + s1[3];
    sq  = s2[0] + s2[1] + s2[2] + s2[3];
    float mean = sum * (1.0f / 512.0f);
    float var  = sq * (1.0f / 512.0f) - mean * mean;
    float inv  = rsqrtf(var + 1e-5f);
#pragma unroll
    for (int i = 0; i < 4; ++i) {
        int idx = tid + i * 128;
        float o = (v[i] - mean) * inv * g[idx] + be[idx];
        if (yf)  yf[(size_t)row * 512 + idx]  = o;
        if (ybf) ybf[(size_t)row * 512 + idx] = f2bf_bits(o);
    }
}

// ---------------------------------------------------------------------------
// Host orchestration
// ---------------------------------------------------------------------------
extern "C" void kernel_launch(void* const* d_in, const int* in_sizes, int n_in,
                              void* d_out, int out_size, void* d_ws, size_t ws_size,
                              hipStream_t stream) {
    (void)in_sizes; (void)n_in; (void)out_size; (void)ws_size;
    const float* x     = (const float*)d_in[0];
    const float* w_qkv = (const float*)d_in[1];
    const float* b_qkv = (const float*)d_in[2];
    const float* w1    = (const float*)d_in[3];
    const float* b1    = (const float*)d_in[4];
    const float* w2    = (const float*)d_in[5];
    const float* b2    = (const float*)d_in[6];
    const float* g1    = (const float*)d_in[7];
    const float* be1   = (const float*)d_in[8];
    const float* g2    = (const float*)d_in[9];
    const float* be2   = (const float*)d_in[10];
    float* out = (float*)d_out;

    char* ws = (char*)d_ws;
    size_t off = 0;
    auto take = [&](size_t bytes) -> void* {
        void* p = (void*)(ws + off);
        off += (bytes + 255) & ~(size_t)255;
        return p;
    };
    const size_t MROWS = 8192;  // B*L
    unsigned short* x_bf  = (unsigned short*)take(MROWS * 512 * 2);
    unsigned short* wqkvT = (unsigned short*)take(1536ull * 512 * 2);
    unsigned short* w1T   = (unsigned short*)take(2048ull * 512 * 2);
    unsigned short* w2T   = (unsigned short*)take(512ull * 2048 * 2);
    unsigned short* q_bf  = (unsigned short*)take(MROWS * 512 * 2);
    unsigned short* k_bf  = (unsigned short*)take(MROWS * 512 * 2);
    unsigned short* vT_bf = (unsigned short*)take(MROWS * 512 * 2);
    float*          attn  = (float*)take(MROWS * 512 * 4);
    float*          yf    = (float*)take(MROWS * 512 * 4);
    unsigned short* y_bf  = (unsigned short*)take(MROWS * 512 * 2);
    float*          ffn   = (float*)take(MROWS * 512 * 4);
    // FFN hidden (8192x2048 bf16 = 32MB) aliases the dead q/k/vT/attn region (40MB):
    unsigned short* h_bf  = q_bf;

    // 0) precision conversion + weight transposes (to [N][K] for B-fragments)
    cvt_bf16_kernel<<<(8192 * 512 + 255) / 256, 256, 0, stream>>>(x, x_bf, 8192 * 512);
    transpose_cvt_kernel<<<(512 * 1536 + 255) / 256, 256, 0, stream>>>(w_qkv, wqkvT, 512, 1536);
    transpose_cvt_kernel<<<(512 * 2048 + 255) / 256, 256, 0, stream>>>(w1, w1T, 512, 2048);
    transpose_cvt_kernel<<<(2048 * 512 + 255) / 256, 256, 0, stream>>>(w2, w2T, 2048, 512);

    // 1) QKV GEMM, scatter into per-head Q(*8)/K and transposed V
    {
        dim3 grid(8192 / 64, 1536 / 64);
        gemm_bf16_kernel<0><<<grid, 128, 0, stream>>>(x_bf, wqkvT, b_qkv, 8192, 1536, 512,
                                                      nullptr, nullptr, q_bf, k_bf, vT_bf);
    }
    // 2) flash attention
    attn_kernel<<<32 * (2048 / 64), 128, 0, stream>>>(q_bf, k_bf, vT_bf, attn);
    // 3) residual + LN1 (f32 out + bf16 mirror for FFN)
    ln_kernel<<<8192, 128, 0, stream>>>(x, attn, g1, be1, yf, y_bf);
    // 4) FFN1 + ReLU
    {
        dim3 grid(8192 / 64, 2048 / 64);
        gemm_bf16_kernel<1><<<grid, 128, 0, stream>>>(y_bf, w1T, b1, 8192, 2048, 512,
                                                      nullptr, h_bf, nullptr, nullptr, nullptr);
    }
    // 5) FFN2
    {
        dim3 grid(8192 / 64, 512 / 64);
        gemm_bf16_kernel<2><<<grid, 128, 0, stream>>>(h_bf, w2T, b2, 8192, 512, 2048,
                                                      ffn, nullptr, nullptr, nullptr, nullptr);
    }
    // 6) residual + LN2 -> output
    ln_kernel<<<8192, 128, 0, stream>>>(yf, ffn, g2, be2, out, nullptr);
}
